// Semantic_Mapping_45646912422340
// MI455X (gfx1250) — compile-verified, matrix operand order checked by reading, not run on hardware
//
#include <hip/hip_runtime.h>
#include <math.h>

typedef __attribute__((ext_vector_type(16))) _Float16 v16h;
typedef __attribute__((ext_vector_type(8)))  float    v8f;

namespace {
constexpr int kBS   = 4;
constexpr int kSH   = 120, kSW = 160;
constexpr int kNPTS = kSH * kSW;            // 19200
constexpr int kNSEM = 16;
constexpr int kVR   = 100;
constexpr int kNSPAT = kVR * kVR;           // 10000
constexpr int kZ    = 80;
constexpr int kZLO  = 13, kZHI = 25, kZR = 12;
constexpr int kMAP  = 240;
constexpr int kMAPSQ = kMAP * kMAP;         // 57600
constexpr int kTILES = kNSPAT / 16;         // 625 (exact)

// workspace layout in floats
constexpr size_t WS_VOX0 = 0;                                           // 4*10000*80    = 3,200,000
constexpr size_t WS_VOXR = (size_t)kBS * kNSPAT * kZ;                   // +4*16*10000*12= 7,680,000
constexpr size_t WS_AVW  = WS_VOXR + (size_t)kBS * kNSEM * kNSPAT * kZR;// +4*20*10000   =   800,000
constexpr size_t WS_POSE = WS_AVW + (size_t)kBS * 20 * kNSPAT;          // +16
constexpr size_t WS_ROT  = 0;   // aliases vox region (vox dead after reduction)

// output layout in floats
constexpr size_t OUT_FPM  = 0;                                  // 4*1*100*100 = 40,000
constexpr size_t OUT_MAP  = 40000;                              // 4*20*240*240 = 4,608,000
constexpr size_t OUT_CUR1 = OUT_MAP + (size_t)kBS * 20 * kMAPSQ;// 4,648,000
constexpr size_t OUT_CUR2 = OUT_CUR1 + 12;
}

// ---------------------------------------------------------------- zero fill (vectorized)
__global__ void sm_zero(float4* __restrict__ p, int n4) {
  int i = blockIdx.x * blockDim.x + threadIdx.x;
  if (i < n4) p[i] = make_float4(0.f, 0.f, 0.f, 0.f);
}

// ---------------------------------------------------------------- pose update
__global__ void sm_pose(const float* __restrict__ pose_obs,
                        const float* __restrict__ poses_last,
                        float* __restrict__ out, float* __restrict__ wsp) {
  int b = threadIdx.x;
  if (b >= kBS) return;
  const float R2D = 57.29577951308232f;
  float px = poses_last[b * 3 + 0], py = poses_last[b * 3 + 1], po = poses_last[b * 3 + 2];
  float r0 = pose_obs[b * 3 + 0], r1 = pose_obs[b * 3 + 1], r2 = pose_obs[b * 3 + 2];
  float t = po / R2D;
  float st = sinf(t), ct = cosf(t);
  float x = px + r0 * ct - r1 * st;
  float y = py + r0 * st + r1 * ct;
  float o = po + r2 * R2D;
  o = fmodf(o - 180.f, 360.f) + 180.f;
  o = fmodf(o + 180.f, 360.f) - 180.f;
  out[OUT_CUR1 + b * 3 + 0] = x; out[OUT_CUR1 + b * 3 + 1] = y; out[OUT_CUR1 + b * 3 + 2] = o;
  out[OUT_CUR2 + b * 3 + 0] = x; out[OUT_CUR2 + b * 3 + 1] = y; out[OUT_CUR2 + b * 3 + 2] = o;
  float t1 = (90.f - o) * 0.017453292519943295f;
  wsp[b * 4 + 0] = cosf(t1);
  wsp[b * 4 + 1] = sinf(t1);
  wsp[b * 4 + 2] = -(x * 20.f - 120.f) / 120.f;   // stx
  wsp[b * 4 + 3] = -(y * 20.f - 120.f) / 120.f;   // sty
}

// ---------------------------------------------------------------- voxel splat (scatter + atomics)
__global__ void sm_splat(const float* __restrict__ obs,
                         float* __restrict__ vox0, float* __restrict__ voxr) {
  int gid = blockIdx.x * blockDim.x + threadIdx.x;
  if (gid >= kBS * kNPTS) return;
  int b = gid / kNPTS, p = gid % kNPTS;
  int h = p / kSW, w = p % kSW;
  const float* ob = obs + (size_t)b * 22 * kNPTS;
  float depth = ob[3 * kNPTS + p];
  const float F = 80.0f / tanf(0.6894050545377601f);   // SW/2 / tan(FOV/2)
  float X  = ((float)w - 79.5f) * depth / F;
  float Zc = ((float)(119 - h) - 59.5f) * depth / F + 88.0f;
  float pos0 = (X + 250.0f) * 0.2f;   // x voxel coord
  float pos1 = depth * 0.2f;          // y (depth) voxel coord
  float pos2 = Zc * 0.2f + 8.0f;      // z voxel coord
  float f0 = floorf(pos0), f1 = floorf(pos1), f2 = floorf(pos2);

  float sem[kNSEM];
#pragma unroll
  for (int c = 0; c < kNSEM; ++c) sem[c] = ob[(4 + c) * kNPTS + p];

  for (int a = 0; a < 2; ++a) {
    float p0 = f0 + (float)a;
    if (!(p0 > 0.f && p0 < 100.f)) continue;
    float w0 = 1.f - fabsf(pos0 - p0);
    for (int bb = 0; bb < 2; ++bb) {
      float p1 = f1 + (float)bb;
      if (!(p1 > 0.f && p1 < 100.f)) continue;
      float w1 = 1.f - fabsf(pos1 - p1);
      for (int cc = 0; cc < 2; ++cc) {
        float p2 = f2 + (float)cc;
        if (!(p2 > 0.f && p2 < 80.f)) continue;
        float w2 = 1.f - fabsf(pos2 - p2);
        float wgt = w0 * w1 * w2;
        if (wgt <= 0.f) continue;
        int s = (int)p0 * 100 + (int)p1;
        int z = (int)p2;
        atomicAdd(&vox0[((size_t)b * kNSPAT + s) * kZ + z], wgt);   // occupancy feat == 1
        if (z >= kZLO && z < kZHI) {
          int zr = z - kZLO;
#pragma unroll
          for (int c = 0; c < kNSEM; ++c)
            atomicAdd(&voxr[(((size_t)(b * kNSEM + c)) * kNSPAT + s) * kZR + zr], wgt * sem[c]);
        }
      }
    }
  }
}

// ---------------------------------------------------------------- WMMA z-reduction
// Sum over z of round(vox) via v_wmma_f32_16x16x32_f16 with an all-ones B.
// D[m,n] = sum_k A[m,k] is invariant to K-slot placement, so each lane loads 16
// consecutive z values of its row (lanes 0-15: K-half 0, lanes 16-31: half 1).
// Loads are UNCONDITIONAL aligned float4 (over-reads stay inside the workspace);
// z-range bounds become v_cndmask selects, never EXEC branches.
__device__ __forceinline__ v8f zsum_chunk(const float* __restrict__ row, int zb,
                                          int zlo, int zhi,
                                          const v16h& ones, v8f acc) {
  const float4* r4 = (const float4*)(row + zb);   // zb is a multiple of 4 -> 16B aligned
  v16h a;
#pragma unroll
  for (int q = 0; q < 4; ++q) {
    float4 f = r4[q];
    float vv[4] = {f.x, f.y, f.z, f.w};
#pragma unroll
    for (int e = 0; e < 4; ++e) {
      int z = zb + q * 4 + e;
      bool in = (z >= zlo) & (z < zhi);
      float v = in ? rintf(vv[e]) : 0.f;          // round-half-even == jnp.round
      a[q * 4 + e] = (_Float16)v;
    }
  }
  return __builtin_amdgcn_wmma_f32_16x16x32_f16(false, a, false, ones,
                                                (short)0, acc, false, false);
}

// D layout: VGPR r, lanes 0-15 -> row r, lanes 16-31 -> row r+8 (cols identical).
__device__ __forceinline__ void store_tile(float* __restrict__ dst, const v8f& v,
                                           int lane, int s0, float invT,
                                           float* __restrict__ fpm) {
  if ((lane & 15) == 0) {
    int hbase = (lane >> 4) * 8;
#pragma unroll
    for (int r = 0; r < 8; ++r) {
      int s = s0 + hbase + r;
      float val = v[r] * invT;
      val = fminf(fmaxf(val, 0.f), 1.f);
      dst[s] = val;
      if (fpm) fpm[(s % 100) * 100 + (s / 100)] = val;   // (py,px) transpose for fp_map
    }
  }
}

__global__ void __launch_bounds__(32)
sm_reduce(const float* __restrict__ vox0, const float* __restrict__ voxr,
          float* __restrict__ avwin, float* __restrict__ fpmap) {
  int b    = blockIdx.x / kTILES;
  int tile = blockIdx.x % kTILES;
  int lane = threadIdx.x;
  int m = lane & 15, hi = lane >> 4;
  int s = tile * 16 + m;             // kNSPAT divisible by 16: always valid
  int h16 = hi * 16;

  v16h ones;
#pragma unroll
  for (int i = 0; i < 16; ++i) ones[i] = (_Float16)1.0f;

  const float* r0 = vox0 + ((size_t)b * kNSPAT + s) * kZ;
  float* av_b = avwin + (size_t)b * 20 * kNSPAT;

  // allp (channel 0, all 80 z) -> fp_exp  (av channel 1)
  v8f acc = {};
  acc = zsum_chunk(r0, 0  + h16, 0, 80, ones, acc);
  acc = zsum_chunk(r0, 32 + h16, 0, 80, ones, acc);
  acc = zsum_chunk(r0, 64 + h16, 0, 80, ones, acc);   // hi=1 half fully masked
  store_tile(av_b + 1 * kNSPAT, acc, lane, tile * 16, 1.0f, nullptr);

  // ahp channel 0 (z 13..24) -> fp_map (av channel 0 + d_out fp_map)
  // window starts at z=12 so the float4 loads stay 16B aligned; mask 13<=z<25.
  v8f acc2 = {};
  acc2 = zsum_chunk(r0, 12 + h16, 13, 25, ones, acc2);
  store_tile(av_b + 0 * kNSPAT, acc2, lane, tile * 16, 1.0f, fpmap + (size_t)b * kNSPAT);

  // av channels 2,3 are zero
  if ((lane & 15) == 0) {
    int hbase = hi * 8;
#pragma unroll
    for (int r = 0; r < 8; ++r) {
      int ss = tile * 16 + hbase + r;
      av_b[2 * kNSPAT + ss] = 0.f;
      av_b[3 * kNSPAT + ss] = 0.f;
    }
  }

  // semantic channels: ahp[1..16] / CAT_T -> av channels 4..19
  for (int c = 0; c < kNSEM; ++c) {
    const float* rr = voxr + (((size_t)(b * kNSEM + c)) * kNSPAT + s) * kZR;
    v8f a3 = {};
    a3 = zsum_chunk(rr, h16, 0, 12, ones, a3);
    store_tile(av_b + (size_t)(4 + c) * kNSPAT, a3, lane, tile * 16, 0.2f, nullptr);
  }
}

// ---------------------------------------------------------------- rotation grid_sample
// av is implicit: zero outside the 100x100 window [120,220)x[70,170); inside,
// value comes from avwin (already clipped), s = (x-70)*100 + (y-120).
__global__ void sm_rotate(const float* __restrict__ avwin, const float* __restrict__ wsp,
                          float* __restrict__ rot) {
  int gid = blockIdx.x * blockDim.x + threadIdx.x;
  if (gid >= kBS * kMAPSQ) return;
  int b = gid / kMAPSQ;
  int ij = gid % kMAPSQ;
  int i = ij / kMAP, j = ij % kMAP;
  float c1 = wsp[b * 4 + 0], s1 = wsp[b * 4 + 1];
  float gx = (2.f * j + 1.f) * (1.0f / kMAP) - 1.f;
  float gy = (2.f * i + 1.f) * (1.0f / kMAP) - 1.f;
  float sx = c1 * gx - s1 * gy;
  float sy = s1 * gx + c1 * gy;
  float GX = (sx + 1.f) * 0.5f * (float)(kMAP - 1);
  float GY = (sy + 1.f) * 0.5f * (float)(kMAP - 1);
  float x0f = floorf(GX), y0f = floorf(GY);
  float wx = GX - x0f, wy = GY - y0f;
  int x0 = (int)x0f, y0 = (int)y0f;
  float wc[4] = {(1.f - wx) * (1.f - wy), wx * (1.f - wy), (1.f - wx) * wy, wx * wy};
  int xc[4] = {x0, x0 + 1, x0, x0 + 1};
  int yc[4] = {y0, y0, y0 + 1, y0 + 1};
  int sidx[4]; float wm[4];
#pragma unroll
  for (int k = 0; k < 4; ++k) {
    bool ok = (xc[k] >= 70) && (xc[k] < 170) && (yc[k] >= 120) && (yc[k] < 220);
    sidx[k] = ok ? ((xc[k] - 70) * 100 + (yc[k] - 120)) : 0;   // safe index when masked
    wm[k] = ok ? wc[k] : 0.f;                                  // masked weight
  }
  const float* av_b = avwin + (size_t)b * 20 * kNSPAT;
  float* rb = rot + (size_t)b * 20 * kMAPSQ + ij;
  for (int c = 0; c < 20; ++c) {
    const float* ac = av_b + (size_t)c * kNSPAT;
    float r = ac[sidx[0]] * wm[0] + ac[sidx[1]] * wm[1]
            + ac[sidx[2]] * wm[2] + ac[sidx[3]] * wm[3];
    rb[(size_t)c * kMAPSQ] = r;
  }
}

// ---------------------------------------------------------------- translation grid_sample + max
__global__ void sm_translate(const float* __restrict__ rot, const float* __restrict__ wsp,
                             const float* __restrict__ maps_last, float* __restrict__ outmap) {
  int gid = blockIdx.x * blockDim.x + threadIdx.x;
  if (gid >= kBS * kMAPSQ) return;
  int b = gid / kMAPSQ;
  int ij = gid % kMAPSQ;
  int i = ij / kMAP, j = ij % kMAP;
  float stx = wsp[b * 4 + 2], sty = wsp[b * 4 + 3];
  float gx = (2.f * j + 1.f) * (1.0f / kMAP) - 1.f + stx;
  float gy = (2.f * i + 1.f) * (1.0f / kMAP) - 1.f + sty;
  float GX = (gx + 1.f) * 0.5f * (float)(kMAP - 1);
  float GY = (gy + 1.f) * 0.5f * (float)(kMAP - 1);
  float x0f = floorf(GX), y0f = floorf(GY);
  float wx = GX - x0f, wy = GY - y0f;
  int x0 = (int)x0f, y0 = (int)y0f;
  float wc[4] = {(1.f - wx) * (1.f - wy), wx * (1.f - wy), (1.f - wx) * wy, wx * wy};
  int xc[4] = {x0, x0 + 1, x0, x0 + 1};
  int yc[4] = {y0, y0, y0 + 1, y0 + 1};
  int midx[4]; float wm[4];
#pragma unroll
  for (int k = 0; k < 4; ++k) {
    bool ok = (xc[k] >= 0) && (xc[k] <= kMAP - 1) && (yc[k] >= 0) && (yc[k] <= kMAP - 1);
    midx[k] = ok ? (yc[k] * kMAP + xc[k]) : 0;
    wm[k] = ok ? wc[k] : 0.f;
  }
  const float* rb = rot + (size_t)b * 20 * kMAPSQ;
  for (int c = 0; c < 20; ++c) {
    const float* rc = rb + (size_t)c * kMAPSQ;
    float v = rc[midx[0]] * wm[0] + rc[midx[1]] * wm[1]
            + rc[midx[2]] * wm[2] + rc[midx[3]] * wm[3];
    size_t o = (size_t)b * 20 * kMAPSQ + (size_t)c * kMAPSQ + ij;
    outmap[o] = fmaxf(maps_last[o], v);
  }
}

// ---------------------------------------------------------------- launch
extern "C" void kernel_launch(void* const* d_in, const int* in_sizes, int n_in,
                              void* d_out, int out_size, void* d_ws, size_t ws_size,
                              hipStream_t stream) {
  (void)in_sizes; (void)n_in; (void)out_size; (void)ws_size;
  const float* obs        = (const float*)d_in[0];
  const float* pose_obs   = (const float*)d_in[1];
  const float* maps_last  = (const float*)d_in[2];
  const float* poses_last = (const float*)d_in[3];
  float* out = (float*)d_out;
  float* ws  = (float*)d_ws;

  float* vox0  = ws + WS_VOX0;
  float* voxr  = ws + WS_VOXR;
  float* avwin = ws + WS_AVW;
  float* wsp   = ws + WS_POSE;
  float* rot   = ws + WS_ROT;      // aliases vox0/voxr (dead after sm_reduce)

  // zero the voxel scatter targets (vox0 + voxr = 10,880,000 floats, /4 for float4)
  {
    int n4 = (int)(WS_AVW / 4);
    sm_zero<<<(n4 + 255) / 256, 256, 0, stream>>>((float4*)ws, n4);
  }
  sm_pose<<<1, 32, 0, stream>>>(pose_obs, poses_last, out, wsp);
  {
    int n = kBS * kNPTS;           // 76,800 points
    sm_splat<<<(n + 255) / 256, 256, 0, stream>>>(obs, vox0, voxr);
  }
  sm_reduce<<<kBS * kTILES, 32, 0, stream>>>(vox0, voxr, avwin, out + OUT_FPM);
  {
    int n = kBS * kMAPSQ;          // 230,400 pixels
    sm_rotate<<<(n + 255) / 256, 256, 0, stream>>>(avwin, wsp, rot);
    sm_translate<<<(n + 255) / 256, 256, 0, stream>>>(rot, wsp, maps_last, out + OUT_MAP);
  }
}
